// EncoderRNN_26671746908744
// MI455X (gfx1250) — matrix operand order repaired and checked by reading, not compiled, not run on hardware
//
#include <hip/hip_runtime.h>
#include <hip/hip_bf16.h>
#include <hip/hip_fp16.h>

#define B_   16
#define L_   256
#define V_   32000
#define E_   256
#define H_   256
#define G4H  1024   // 4*H

typedef __attribute__((ext_vector_type(16))) _Float16 v16h;
typedef __attribute__((ext_vector_type(8)))  _Float16 v8h;
typedef __attribute__((ext_vector_type(8)))  float    v8f;

__device__ __forceinline__ float fast_sigmoid(float x) {
    return 1.f / (1.f + __expf(-x));
}
__device__ __forceinline__ float fast_tanh(float x) {
    float e = __expf(2.f * x);
    return (e - 1.f) / (e + 1.f);
}

// ---------------------------------------------------------------------------
// Kernel 1: one block per token. Stream the one-hot row (float4), find the
// nonzero index, gather the embedding row, store as f16 in time-major layout
// x16[(l*B + b) * E + e].  This is the HBM-bound stage (~524 MB read).
// ---------------------------------------------------------------------------
__global__ void embed_gather(const float* __restrict__ seq,
                             const float* __restrict__ emb,
                             _Float16* __restrict__ x16) {
    int tok = blockIdx.x;              // tok = b*L + l
    int b = tok / L_, l = tok % L_;
    __shared__ int s_id;
    if (threadIdx.x == 0) s_id = 0;
    __syncthreads();
    const float4* row4 = (const float4*)(seq + (size_t)tok * V_);
    for (int i = threadIdx.x; i < V_ / 4; i += blockDim.x) {
        float4 v = row4[i];
        if (v.x != 0.f || v.y != 0.f || v.z != 0.f || v.w != 0.f) {
            int off = (v.x != 0.f) ? 0 : (v.y != 0.f) ? 1 : (v.z != 0.f) ? 2 : 3;
            s_id = i * 4 + off;        // exactly one nonzero per row
        }
    }
    __syncthreads();
    const float* erow = emb + (size_t)s_id * E_;
    _Float16* xr = x16 + ((size_t)l * B_ + b) * E_;
    for (int e = threadIdx.x; e < E_; e += blockDim.x)
        xr[e] = (_Float16)erow[e];
}

// ---------------------------------------------------------------------------
// Kernel 2: fp32 -> fp16 weight conversion (keeps weights hot in L2 as f16).
// ---------------------------------------------------------------------------
__global__ void cvt_f32_f16(const float* __restrict__ src,
                            _Float16* __restrict__ dst, int n) {
    int i = blockIdx.x * blockDim.x + threadIdx.x;
    if (i < n) dst[i] = (_Float16)src[i];
}

// ---------------------------------------------------------------------------
// Kernel 3: pre-gates for ALL timesteps: gpre[dir][t*16+b][n] = x@W_ih^T + bias
// One wave per 16x16 output tile, K-loop of 8 v_wmma_f32_16x16x32_f16.
// Bias folded in as the C initializer (C layout: N = lane%16 for all 8 rows).
// ---------------------------------------------------------------------------
__global__ void gemm_xw(const _Float16* __restrict__ x16,   // [L*B][E]
                        const _Float16* __restrict__ wih16, // [2][4H][E]
                        const float* __restrict__ bias_f,
                        const float* __restrict__ bias_b,
                        float* __restrict__ gpre) {         // [2][L*16][4H]
    int w    = blockIdx.x * (blockDim.x >> 5) + (threadIdx.x >> 5);
    int lane = threadIdx.x & 31;
    const int MT = L_;            // 256 m-tiles (one per timestep, rows = batch)
    const int NT = G4H / 16;      // 64 n-tiles
    int dir = w / (MT * NT);
    int rem = w % (MT * NT);
    int tm  = rem / NT;
    int tn  = rem % NT;

    const _Float16* W    = wih16 + (size_t)dir * G4H * E_;
    const float*    bias = dir ? bias_b : bias_f;

    int n  = tn * 16 + (lane & 15);
    float bn = bias[n];
    v8f acc = {bn, bn, bn, bn, bn, bn, bn, bn};

    int mrow = tm * 16 + (lane & 15);
    const _Float16* arow = x16 + (size_t)mrow * E_;
    const _Float16* brow = W + (size_t)n * E_;
    int a_koff = (lane < 16) ? 0 : 8;    // A 16x32 f16 per-lane K offsets
    int b_koff = (lane < 16) ? 0 : 8;    // B 32x16 f16 per-lane K offsets

#pragma unroll
    for (int kk = 0; kk < E_ / 32; ++kk) {
        union { v16h v; v8h h[2]; } a, bm;
        int ka = kk * 32 + a_koff;
        a.h[0]  = *(const v8h*)(arow + ka);
        a.h[1]  = *(const v8h*)(arow + ka + 16);
        int kb = kk * 32 + b_koff;
        bm.h[0] = *(const v8h*)(brow + kb);
        bm.h[1] = *(const v8h*)(brow + kb + 16);
        acc = __builtin_amdgcn_wmma_f32_16x16x32_f16(false, a.v, false, bm.v,
                                                     (short)0, acc, false, false);
    }

    float* gp = gpre + ((size_t)dir * MT * 16 + (size_t)tm * 16) * G4H;
    int mbase = (lane < 16) ? 0 : 8;
#pragma unroll
    for (int v = 0; v < 8; ++v)
        gp[(size_t)(mbase + v) * G4H + n] = acc[v];
}

// ---------------------------------------------------------------------------
// Kernel 4: persistent LSTM recurrence. One 1024-thread block per direction.
// h (f16, 8KB) and gates (f16, 32KB) live in LDS; W_hh re-streamed from L2.
// Per step, per wave: 2 N-tiles x 8 K-steps of v_wmma_f32_16x16x32_f16, then
// a fused elementwise phase (sigmoid/tanh, cell update, length masking).
// ---------------------------------------------------------------------------
__global__ void __launch_bounds__(1024)
lstm_persistent(const _Float16* __restrict__ whh16, // [2][4H][H]
                const float* __restrict__ gpre,     // [2][L*16][4H]
                const int* __restrict__ lengths,    // [B]
                float* __restrict__ out) {          // [B][L][2H]
    int dir  = blockIdx.x;
    int tid  = threadIdx.x;
    int wave = tid >> 5, lane = tid & 31;

    __shared__ _Float16 sh[B_ * H_];       // hidden state h[b][j]
    __shared__ _Float16 sgate[B_ * G4H];   // gate pre-activations

    const _Float16* W = whh16 + (size_t)dir * G4H * H_;
    int nbase0 = wave * 32;                // this wave owns N in [nbase0, nbase0+32)

    // elementwise ownership: 4 consecutive units (b, j..j+3) per thread
    int g  = tid * 4;
    int eb = g >> 8;
    int ej = g & 255;
    int mylen = lengths[eb];
    float c_st[4] = {0.f, 0.f, 0.f, 0.f};

    for (int i = tid; i < B_ * H_; i += blockDim.x) sh[i] = (_Float16)0.f;
    __syncthreads();

    const float* gp_dir = gpre + (size_t)dir * L_ * 16 * G4H;
    int mbase  = (lane < 16) ? 0 : 8;
    int a_koff = (lane < 16) ? 0 : 8;
    int b_koff = (lane < 16) ? 0 : 8;
    int arow   = lane & 15;                // batch row for A fragment

    for (int s = 0; s < L_; ++s) {
        int t = dir ? (L_ - 1 - s) : s;
        const float* gp = gp_dir + (size_t)t * 16 * G4H;

        // prefetch next step's pre-gates into L2/L0
        if (s + 1 < L_) {
            int tn = dir ? (t - 1) : (t + 1);
            __builtin_prefetch(gp_dir + (size_t)tn * 16 * G4H + tid * 16, 0, 1);
        }

        // ---- gates = h @ W_hh^T + gpre[t] ----
#pragma unroll
        for (int tt = 0; tt < 2; ++tt) {
            int n = nbase0 + tt * 16 + (lane & 15);
            const _Float16* brow = W + (size_t)n * H_;
            v8f acc;
#pragma unroll
            for (int v = 0; v < 8; ++v)
                acc[v] = gp[(size_t)(mbase + v) * G4H + n];
#pragma unroll
            for (int kk = 0; kk < 8; ++kk) {
                union { v16h v; v8h h[2]; } a, bm;
                int ka = kk * 32 + a_koff;
                const _Float16* hr = sh + (size_t)arow * H_;
                a.h[0]  = *(const v8h*)(hr + ka);
                a.h[1]  = *(const v8h*)(hr + ka + 16);
                int kb = kk * 32 + b_koff;
                bm.h[0] = *(const v8h*)(brow + kb);
                bm.h[1] = *(const v8h*)(brow + kb + 16);
                acc = __builtin_amdgcn_wmma_f32_16x16x32_f16(false, a.v, false, bm.v,
                                                             (short)0, acc, false, false);
            }
#pragma unroll
            for (int v = 0; v < 8; ++v)
                sgate[(size_t)(mbase + v) * G4H + n] = (_Float16)acc[v];
        }
        __syncthreads();

        // ---- elementwise: gates -> (c, h), masked by sequence length ----
        bool m = (t < mylen);
#pragma unroll
        for (int u = 0; u < 4; ++u) {
            int j = ej + u;
            float ig = fast_sigmoid((float)sgate[eb * G4H + j]);
            float fg = fast_sigmoid((float)sgate[eb * G4H + H_ + j]);
            float gg = fast_tanh   ((float)sgate[eb * G4H + 2 * H_ + j]);
            float og = fast_sigmoid((float)sgate[eb * G4H + 3 * H_ + j]);
            float cn = fg * c_st[u] + ig * gg;
            float hn = og * fast_tanh(cn);
            float hp = (float)sh[eb * H_ + j];
            c_st[u] = m ? cn : c_st[u];
            sh[eb * H_ + j] = (_Float16)(m ? hn : hp);
            out[((size_t)eb * L_ + t) * (2 * H_) + (size_t)dir * H_ + j] = m ? hn : 0.f;
        }
        __syncthreads();
    }
}

// ---------------------------------------------------------------------------
extern "C" void kernel_launch(void* const* d_in, const int* in_sizes, int n_in,
                              void* d_out, int out_size, void* d_ws, size_t ws_size,
                              hipStream_t stream) {
    const float* seq    = (const float*)d_in[0];
    const float* emb    = (const float*)d_in[1];
    const float* W_ih_f = (const float*)d_in[2];
    const float* W_hh_f = (const float*)d_in[3];
    const float* b_f    = (const float*)d_in[4];
    const float* W_ih_b = (const float*)d_in[5];
    const float* W_hh_b = (const float*)d_in[6];
    const float* b_b    = (const float*)d_in[7];
    const int*   lens   = (const int*)d_in[8];
    float* out = (float*)d_out;

    char* ws = (char*)d_ws;
    _Float16* x16   = (_Float16*)(ws);                        // 2 MB: [L*B][E]
    _Float16* wih16 = (_Float16*)(ws + (2ull << 20));         // 1 MB: [2][4H][E]
    _Float16* whh16 = (_Float16*)(ws + (3ull << 20));         // 1 MB: [2][4H][H]
    float*    gpre  = (float*)   (ws + (4ull << 20));         // 32 MB: [2][L*16][4H]

    // 1) embedding gather (one block per token, streams 524 MB of one-hot)
    embed_gather<<<B_ * L_, 256, 0, stream>>>(seq, emb, x16);

    // 2) weight conversion to f16
    const int WN = G4H * E_;                                  // 262144
    cvt_f32_f16<<<WN / 256, 256, 0, stream>>>(W_ih_f, wih16,            WN);
    cvt_f32_f16<<<WN / 256, 256, 0, stream>>>(W_ih_b, wih16 + WN,       WN);
    cvt_f32_f16<<<WN / 256, 256, 0, stream>>>(W_hh_f, whh16,            WN);
    cvt_f32_f16<<<WN / 256, 256, 0, stream>>>(W_hh_b, whh16 + WN,       WN);

    // 3) pre-gates GEMM: 2 dirs x 256 m-tiles x 64 n-tiles = 32768 waves
    gemm_xw<<<(2 * L_ * (G4H / 16)) / 8, 256, 0, stream>>>(x16, wih16, b_f, b_b, gpre);

    // 4) persistent bidirectional recurrence: one WGP-sized block per direction
    lstm_persistent<<<2, 1024, 0, stream>>>(whh16, gpre, lens, out);
}
